// ImprovedFullNetwork_74277164417472
// MI455X (gfx1250) — matrix operand order repaired and checked by reading, not compiled, not run on hardware
//
#include <hip/hip_runtime.h>
#include <math.h>

// ---------------- problem constants ----------------
constexpr int   BPATHS  = 262144;
constexpr int   DD      = 5;
constexpr int   NSTEPS  = 20;
constexpr float HH      = 0.1f;
constexpr float RR      = 0.06f;
constexpr float TT      = 2.0f;
constexpr float SIG     = 0.2f;
constexpr float BDRIFT  = 0.1f;
constexpr float CURTAGE = 0.005f;
constexpr int   NBINS   = 4096;
constexpr int   CHUNK   = 256 * DD;      // floats of dW per block per step (=1280, 5KB)

typedef __attribute__((ext_vector_type(16))) _Float16     v16h;
typedef __attribute__((ext_vector_type(8)))  _Float16     v8h;
typedef __attribute__((ext_vector_type(8)))  float        v8f;
typedef __attribute__((ext_vector_type(4)))  unsigned int u32x4;
typedef __attribute__((ext_vector_type(8)))  int          i32x8;
typedef __attribute__((ext_vector_type(4)))  int          i32x4;

union AFrag { v16h v; v8h h[2]; _Float16 e[16]; };

#if __has_builtin(__builtin_amdgcn_tensor_load_to_lds) && __has_builtin(__builtin_amdgcn_s_wait_tensorcnt)
#define HAVE_TDM 1
#else
#define HAVE_TDM 0
#endif

#if HAVE_TDM
// One TDM DMA: contiguous CHUNK floats from global -> LDS.
// D# group0: count=1 | lds_addr | 57b global addr | type=2 (bits 127:126)
// D# group1: data_size=4B, tensor_dim0=tile_dim0=1280, tensor_dim1=1, stride0=1280
__device__ __forceinline__ void tdm_load_chunk(const float* gptr, void* ldsptr) {
    unsigned           lds = (unsigned)(unsigned long long)(size_t)ldsptr;
    unsigned long long ga  = (unsigned long long)(size_t)gptr;
    u32x4 g0 = {1u, lds, (unsigned)ga,
                (unsigned)((ga >> 32) & 0x01FFFFFFull) | 0x80000000u};
    i32x8 g1 = {0x20000,                        // data_size=2 (4B) at [17:16]
                (int)((CHUNK & 0xFFFF) << 16),  // tensor_dim0 lo16 at [63:48]
                0x10000,                        // tensor_dim1=1 at [95:80]
                (int)((CHUNK & 0xFFFF) << 16),  // tile_dim0 at [127:112]
                0,                              // tile_dim1=0, tile_dim2=0
                CHUNK,                          // tensor_dim0_stride lo32
                0, 0};
    i32x4 gz = {0, 0, 0, 0};
#if defined(__clang_major__) && __clang_major__ >= 23
    i32x8 g4 = {0, 0, 0, 0, 0, 0, 0, 0};
    __builtin_amdgcn_tensor_load_to_lds(g0, g1, gz, gz, g4, 0);
#else
    __builtin_amdgcn_tensor_load_to_lds(g0, g1, gz, gz, 0);
#endif
}
#endif

// monotone float <-> uint key (for atomic min/max)
__device__ __forceinline__ unsigned fkey(float f) {
    unsigned u = __float_as_uint(f);
    return (u & 0x80000000u) ? ~u : (u | 0x80000000u);
}
__device__ __forceinline__ float fdec(unsigned k) {
    unsigned u = (k & 0x80000000u) ? (k & 0x7fffffffu) : ~k;
    return __uint_as_float(u);
}

// stats layout (floats at ws + BPATHS):
// 0 sum_x | 1 sum_x2 | 2 sum_cT | 3 sum_c0 | 4 sum_alpha | 5 K01
// 6 t5 | 7 t95 | 8 lo_sum | 9 lo_cnt | 10 hi_sum | 11 hi_cnt
// (as uint) 12 minkey | 13 maxkey ; hist (uint[4096]) at ws + BPATHS + 16

// ---------------- init ----------------
__global__ void init_kernel(float* stats, unsigned* hist) {
    int tid = threadIdx.x;
    unsigned* su = (unsigned*)stats;
    if (tid < 16) {
        if (tid == 12)      su[12] = 0xFFFFFFFFu;
        else if (tid == 13) su[13] = 0u;
        else                stats[tid] = 0.0f;
    }
    for (int i = tid; i < NBINS; i += blockDim.x) hist[i] = 0u;
}

// ---------------- rollout: TDM-fed, per-lane path state + WMMA MLP ----------------
__global__ void __launch_bounds__(256)
rollout_kernel(const float* __restrict__ x_in, const float* __restrict__ dW,
               const float* __restrict__ Wo,  const float* __restrict__ bo,
               const float* __restrict__ Wk,  const float* __restrict__ bk,
               const float* __restrict__ Wu0, const float* __restrict__ bu0,
               const float* __restrict__ W1g, const float* __restrict__ b1g,
               const float* __restrict__ W2g, const float* __restrict__ b2g,
               const float* __restrict__ W3g, const float* __restrict__ b3g,
               float* __restrict__ xarr, float* __restrict__ stats) {
    const int tid  = threadIdx.x;
    const int lane = tid & 31;
    const int wv   = tid >> 5;          // wave id (8 waves/block)
    const int hlf  = lane >> 4;         // WMMA K-group select
    const int mr   = lane & 15;         // row/col within 16-tile
    const int p    = blockIdx.x * 256 + tid;
    const int p0   = blockIdx.x * 256;

    // ---- LDS (65,216 B total) ----
    __shared__ __align__(16) _Float16 sW2T[NSTEPS - 1][20][32]; // [s][outCol][k]
    __shared__ __align__(16) _Float16 sW3T[NSTEPS - 1][DD][32];
    __shared__ __align__(16) _Float16 zrow[32];                 // zero B-frag row
    __shared__ float sW1[NSTEPS - 1][20], sB1[NSTEPS - 1][20];
    __shared__ float sB2[NSTEPS - 1][32];                       // bias padded to 32
    __shared__ float sB3[NSTEPS - 1][8];
    __shared__ float xsh[8][32];
    __shared__ __align__(16) _Float16 h2st[8][16][32];          // one M-tile stage
    __shared__ float ast[8][32][8];
    __shared__ float red[256];
    __shared__ __align__(16) float dwbuf[2][CHUNK];             // TDM double buffer

    // kick off DMA of step-0 noise before anything else (overlaps weight preload)
#if HAVE_TDM
    if (wv == 0) tdm_load_chunk(dW + (size_t)p0 * DD, (void*)&dwbuf[0][0]);
#else
    {
        const float* g = dW + (size_t)p0 * DD;
        for (int i = tid; i < CHUNK; i += 256) dwbuf[0][i] = g[i];
    }
#endif

    // cooperative weight preload (+ f32->f16, K pad 20->32 zeroed)
    for (int i = tid; i < (NSTEPS - 1) * 20 * 32; i += 256) {
        int s = i / (20 * 32), r = i % (20 * 32), col = r / 32, k = r % 32;
        sW2T[s][col][k] = (k < 20) ? (_Float16)W2g[s * 400 + k * 20 + col] : (_Float16)0.0f;
    }
    for (int i = tid; i < (NSTEPS - 1) * DD * 32; i += 256) {
        int s = i / (DD * 32), r = i % (DD * 32), c = r / 32, k = r % 32;
        sW3T[s][c][k] = (k < 20) ? (_Float16)W3g[s * 100 + k * DD + c] : (_Float16)0.0f;
    }
    for (int i = tid; i < (NSTEPS - 1) * 20; i += 256) {
        int s = i / 20, k = i % 20;
        sW1[s][k] = W1g[i]; sB1[s][k] = b1g[i];
    }
    for (int i = tid; i < (NSTEPS - 1) * 32; i += 256) {
        int s = i / 32, k = i % 32;
        sB2[s][k] = (k < 20) ? b2g[s * 20 + k] : 0.0f;
    }
    for (int i = tid; i < (NSTEPS - 1) * 8; i += 256) {
        int s = i / 8, j = i % 8;
        sB3[s][j] = (j < 5) ? b3g[s * 5 + j] : 0.0f;
    }
    if (tid < 32) zrow[tid] = (_Float16)0.0f;

    // ---- per-path heads ----
    const float x0 = x_in[p];
    float x = x0;

    float alpha[DD];
#pragma unroll
    for (int d = 0; d < DD; ++d) alpha[d] = fmaf(x0, Wu0[d], bu0[d]);

    float lg[11], mxl = -1e30f;
#pragma unroll
    for (int j = 0; j < 11; ++j) { lg[j] = fmaf(x0, Wo[j], bo[j]); mxl = fmaxf(mxl, lg[j]); }
    float Z = 0.0f;
#pragma unroll
    for (int j = 0; j < 11; ++j) Z += __expf(lg[j] - mxl);
    const float asum = (Z - __expf(lg[10] - mxl)) / Z;

    const float K0  = 1.0f + 0.25f * tanhf(fmaf(x0, Wk[0], bk[0]));
    const float K01 = 1.0f + 0.25f * tanhf(fmaf(x0, Wk[1], bk[1]));

    float S[DD], SQ[DD], Nm1[DD], dN[DD];
#pragma unroll
    for (int d = 0; d < DD; ++d) { S[d] = 1.0f; SQ[d] = 1.0f; Nm1[d] = 0.0f; dN[d] = 0.0f; }

    // ---- Euler-Maruyama rollout (TDM double-buffered noise) ----
    for (int n = 1; n <= NSTEPS; ++n) {
        // prefetch next step's chunk, then make current chunk visible
#if HAVE_TDM
        if (wv == 0) {
            if (n < NSTEPS) {
                tdm_load_chunk(dW + ((size_t)n * BPATHS + p0) * DD,
                               (void*)&dwbuf[n & 1][0]);
                __builtin_amdgcn_s_wait_tensorcnt(1);   // current chunk landed
            } else {
                __builtin_amdgcn_s_wait_tensorcnt(0);
            }
        }
#else
        if (n < NSTEPS) {
            const float* g = dW + ((size_t)n * BPATHS + p0) * DD;
            for (int i = tid; i < CHUNK; i += 256) dwbuf[n & 1][i] = g[i];
        }
#endif
        __syncthreads();

        const float* dwp = &dwbuf[(n - 1) & 1][tid * DD];
        float dw[DD];
#pragma unroll
        for (int d = 0; d < DD; ++d) dw[d] = dwp[d];

        float usum = 0.0f, drift = 0.0f;
#pragma unroll
        for (int d = 0; d < DD; ++d) {
            float u  = alpha[d];
            float Np = u / S[d];
            if (n > 1) dN[d] += fabsf(Np - Nm1[d]);
            Nm1[d] = Np;
            usum  += u;
            float gP = BDRIFT * HH + SIG * dw[d];
            float gQ = RR * HH + SIG * dw[d];
            drift += u * gP;
            S[d]  *= (1.0f + gP);
            SQ[d] *= (1.0f + gQ);
        }
        x = x + (x - usum) * (RR * HH) + drift;

        if (n < NSTEPS) {
            // -------- MLP via WMMA: x -> 20 -> 20 -> 5 --------
            const int s = n - 1;
            xsh[wv][lane] = x;
            __syncthreads();

            // layer-3 B/C fragments (shared by both M-tiles); zero-row select,
            // no EXEC divergence around the loads
            AFrag b3f;
            {
                const v8h* wp8 = (const v8h*)((mr < 5) ? &sW3T[s][mr][0] : &zrow[0]);
                b3f.h[0] = wp8[hlf]; b3f.h[1] = wp8[2 + hlf];
            }
            const float cb3 = sB3[s][min(mr, 7)];   // entries 5..7 are zero
            v8f c3 = {cb3, cb3, cb3, cb3, cb3, cb3, cb3, cb3};

#pragma unroll
            for (int t = 0; t < 2; ++t) {
                // layer 1: outer product -> A frag (VALU)
                float xm = xsh[wv][t * 16 + mr];
                AFrag a1;
#pragma unroll
                for (int v = 0; v < 8; ++v) {
                    int k0 = (v < 4 ? 2 * v : 16 + 2 * (v - 4)) + (hlf ? 8 : 0);
#pragma unroll
                    for (int j = 0; j < 2; ++j) {
                        int k = k0 + j;
                        float hv = 0.0f;
                        if (k < 20) hv = fmaxf(fmaf(xm, sW1[s][k], sB1[s][k]), 0.0f);
                        a1.e[2 * v + j] = (_Float16)hv;
                    }
                }
                // layer 2: two N-tiles of WMMA (zero-row select for padded cols)
#pragma unroll
                for (int nt = 0; nt < 2; ++nt) {
                    int col = nt * 16 + mr;
                    AFrag bf;
                    const v8h* wp8 =
                        (const v8h*)((col < 20) ? &sW2T[s][col][0] : &zrow[0]);
                    bf.h[0] = wp8[hlf]; bf.h[1] = wp8[2 + hlf];
                    const float cb = sB2[s][col];   // padded, cols 20..31 zero
                    v8f c = {cb, cb, cb, cb, cb, cb, cb, cb};
                    v8f acc = __builtin_amdgcn_wmma_f32_16x16x32_f16(
                        false, a1.v, false, bf.v, (short)0, c, false, false);
#pragma unroll
                    for (int r = 0; r < 8; ++r) {
                        int m = r + (hlf ? 8 : 0);
                        h2st[wv][m][col] = (_Float16)fmaxf(acc[r], 0.0f);
                    }
                }
                __syncthreads();

                // layer 3: one WMMA (N=5 valid)
                AFrag a2;
                const v8h* r8 = (const v8h*)&h2st[wv][mr][0];
                a2.h[0] = r8[hlf]; a2.h[1] = r8[2 + hlf];
                v8f o = __builtin_amdgcn_wmma_f32_16x16x32_f16(
                    false, a2.v, false, b3f.v, (short)0, c3, false, false);
                if (mr < 5) {
#pragma unroll
                    for (int r = 0; r < 8; ++r)
                        ast[wv][t * 16 + r + (hlf ? 8 : 0)][mr] = o[r];
                }
                __syncthreads();
            }
#pragma unroll
            for (int d = 0; d < DD; ++d) alpha[d] = ast[wv][lane][d];
        }
    }

    // ---- per-path epilogue + block reduction ----
    float dNsum = 0.0f;
#pragma unroll
    for (int d = 0; d < DD; ++d) dNsum += dN[d];
    const float xfin = x - CURTAGE * dNsum;
    xarr[p] = xfin;
    if (p == 0) stats[5] = K01;

    float vals[5] = {xfin, xfin * xfin,
                     fmaxf(S[0] - K0, 0.0f), fmaxf(SQ[0] - K0, 0.0f), asum};
#pragma unroll
    for (int q = 0; q < 5; ++q) {
        red[tid] = vals[q];
        __syncthreads();
        for (int s2 = 128; s2 > 0; s2 >>= 1) {
            if (tid < s2) red[tid] += red[tid + s2];
            __syncthreads();
        }
        if (tid == 0) atomicAdd(&stats[q], red[0]);
        __syncthreads();
    }
    unsigned* redu = (unsigned*)red;
    unsigned* su   = (unsigned*)stats;
    redu[tid] = fkey(xfin);
    __syncthreads();
    for (int s2 = 128; s2 > 0; s2 >>= 1) {
        if (tid < s2) redu[tid] = min(redu[tid], redu[tid + s2]);
        __syncthreads();
    }
    if (tid == 0) atomicMin(&su[12], redu[0]);
    __syncthreads();
    redu[tid] = fkey(xfin);
    __syncthreads();
    for (int s2 = 128; s2 > 0; s2 >>= 1) {
        if (tid < s2) redu[tid] = max(redu[tid], redu[tid + s2]);
        __syncthreads();
    }
    if (tid == 0) atomicMax(&su[13], redu[0]);
}

// ---------------- histogram over final wealth ----------------
__global__ void hist_kernel(const float* __restrict__ xarr, float* stats,
                            unsigned* __restrict__ hist) {
    const unsigned* su = (const unsigned*)stats;
    const float mn = fdec(su[12]);
    const float mx = fdec(su[13]);
    const float scale = (float)NBINS / fmaxf(mx - mn, 1e-12f);

    __shared__ unsigned lh[NBINS];
    for (int i = threadIdx.x; i < NBINS; i += blockDim.x) lh[i] = 0u;
    __syncthreads();
    const int stride = gridDim.x * blockDim.x;
    for (int i = blockIdx.x * blockDim.x + threadIdx.x; i < BPATHS; i += stride) {
        int b = (int)((xarr[i] - mn) * scale);
        b = min(max(b, 0), NBINS - 1);
        atomicAdd(&lh[b], 1u);
    }
    __syncthreads();
    for (int i = threadIdx.x; i < NBINS; i += blockDim.x)
        if (lh[i]) atomicAdd(&hist[i], lh[i]);
}

// ---------------- thresholds: approximate p5 / p95 from histogram ----------------
__global__ void thresh_kernel(float* stats, const unsigned* __restrict__ hist) {
    if (threadIdx.x != 0 || blockIdx.x != 0) return;
    const unsigned* su = (const unsigned*)stats;
    const float mn = fdec(su[12]);
    const float mx = fdec(su[13]);
    const float wbin = (mx - mn) / (float)NBINS;
    const float r5  = 0.05f * (float)(BPATHS - 1);
    const float r95 = 0.95f * (float)(BPATHS - 1);
    float t5 = mn, t95 = mx;
    float cum = 0.0f;
    bool f5 = false, f95 = false;
    for (int b = 0; b < NBINS; ++b) {
        float cnt = (float)hist[b];
        if (!f5 && cum + cnt > r5) {
            t5 = mn + ((float)b + (r5 - cum) / fmaxf(cnt, 1.0f)) * wbin;
            f5 = true;
        }
        if (!f95 && cum + cnt > r95) {
            t95 = mn + ((float)b + (r95 - cum) / fmaxf(cnt, 1.0f)) * wbin;
            f95 = true;
        }
        cum += cnt;
    }
    stats[6] = t5;
    stats[7] = t95;
}

// ---------------- tail sums (CVaR numerators) ----------------
__global__ void tails_kernel(const float* __restrict__ xarr, float* stats) {
    const float t5  = stats[6];
    const float t95 = stats[7];
    float ls = 0.0f, lc = 0.0f, hs = 0.0f, hc = 0.0f;
    const int stride = gridDim.x * blockDim.x;
    for (int i = blockIdx.x * blockDim.x + threadIdx.x; i < BPATHS; i += stride) {
        float xv = xarr[i];
        if (xv < t5)  { ls += xv; lc += 1.0f; }
        if (xv > t95) { hs += xv; hc += 1.0f; }
    }
    __shared__ float red[256];
    float vals[4] = {ls, lc, hs, hc};
    const int tid = threadIdx.x;
#pragma unroll
    for (int q = 0; q < 4; ++q) {
        red[tid] = vals[q];
        __syncthreads();
        for (int s2 = 128; s2 > 0; s2 >>= 1) {
            if (tid < s2) red[tid] += red[tid + s2];
            __syncthreads();
        }
        if (tid == 0) atomicAdd(&stats[8 + q], red[0]);
        __syncthreads();
    }
}

// ---------------- finalize: 7 scalar losses ----------------
__global__ void finalize_kernel(const float* __restrict__ stats, float* __restrict__ out) {
    if (threadIdx.x != 0 || blockIdx.x != 0) return;
    const float Bf   = (float)BPATHS;
    const float mean = stats[0] / Bf;
    const float var  = stats[1] / Bf - mean * mean;
    const float clo  = fmaxf(stats[9],  1.0f);
    const float chi  = fmaxf(stats[11], 1.0f);
    const float disc = expf(-RR * TT);
    const float C00  = disc * stats[3] / Bf;
    out[0] = -mean;
    out[1] = var;
    out[2] = -stats[8]  / clo;
    out[3] = -stats[10] / chi;
    out[4] = (stats[2] / Bf) / (C00 + 1e-8f);
    out[5] = stats[5];
    out[6] = 1.0f - stats[4] / Bf;
}

// ---------------- host launch ----------------
extern "C" void kernel_launch(void* const* d_in, const int* in_sizes, int n_in,
                              void* d_out, int out_size, void* d_ws, size_t ws_size,
                              hipStream_t stream) {
    const float* x_in = (const float*)d_in[0];
    const float* dW   = (const float*)d_in[1];
    const float* Wo   = (const float*)d_in[2];
    const float* bo   = (const float*)d_in[3];
    const float* Wk   = (const float*)d_in[4];
    const float* bk   = (const float*)d_in[5];
    const float* Wu0  = (const float*)d_in[6];
    const float* bu0  = (const float*)d_in[7];
    const float* W1   = (const float*)d_in[8];
    const float* b1   = (const float*)d_in[9];
    const float* W2   = (const float*)d_in[10];
    const float* b2   = (const float*)d_in[11];
    const float* W3   = (const float*)d_in[12];
    const float* b3   = (const float*)d_in[13];

    float*    wsf   = (float*)d_ws;
    float*    xarr  = wsf;                       // [BPATHS]
    float*    stats = wsf + BPATHS;              // 16 floats (12/13 as uint keys)
    unsigned* hist  = (unsigned*)(wsf + BPATHS + 16);
    float*    out   = (float*)d_out;

    init_kernel<<<1, 256, 0, stream>>>(stats, hist);
    rollout_kernel<<<BPATHS / 256, 256, 0, stream>>>(
        x_in, dW, Wo, bo, Wk, bk, Wu0, bu0, W1, b1, W2, b2, W3, b3, xarr, stats);
    hist_kernel<<<256, 256, 0, stream>>>(xarr, stats, hist);
    thresh_kernel<<<1, 32, 0, stream>>>(stats, hist);
    tails_kernel<<<256, 256, 0, stream>>>(xarr, stats);
    finalize_kernel<<<1, 32, 0, stream>>>(stats, out);
    (void)in_sizes; (void)n_in; (void)out_size; (void)ws_size;
}
